// MultiHeadAttention_80831284510748
// MI455X (gfx1250) — matrix operand order repaired and checked by reading, not compiled, not run on hardware
//
#include <hip/hip_runtime.h>
#include <hip/hip_bf16.h>

typedef __bf16 bf16_t;
typedef __attribute__((ext_vector_type(16))) __bf16 v16bf;
typedef __attribute__((ext_vector_type(8)))  __bf16 v8bf;
typedef __attribute__((ext_vector_type(4)))  __bf16 v4bf;
typedef __attribute__((ext_vector_type(8)))  float  v8f;

#define D_MODEL 1024
#define SEQ     2048
#define BATCH   4
#define NHEADS  16
#define DKHEAD  64
#define MROWS   (BATCH * SEQ)

// ---------- WMMA fragment helpers (bf16, 16x16x32) ----------
// A operand (16x32, row-major source): lane m = lane&15 holds row m.
// Lanes 0-15 hold K {0..7, 16..23}; lanes 16-31 hold K {8..15, 24..31}.
// -> two contiguous 16-byte chunks at (base+khalf) and (base+16+khalf), khalf = 8*(lane>=16).
// B operand (32x16): lane n = lane&15 holds column n (a contiguous row of W / K / vT).
// Lanes 0-15 hold K {0..7, 16..23}; lanes 16-31 hold K {8..15, 24..31} (same as A).
__device__ __forceinline__ v16bf make_frag(const bf16_t* p0, const bf16_t* p1) {
  v8bf lo = *reinterpret_cast<const v8bf*>(p0);
  v8bf hi = *reinterpret_cast<const v8bf*>(p1);
  v16bf r;
#pragma unroll
  for (int i = 0; i < 8; ++i) { r[i] = lo[i]; r[i + 8] = hi[i]; }
  return r;
}
__device__ __forceinline__ v16bf fragA(const bf16_t* base, int khalf) {
  return make_frag(base + khalf, base + 16 + khalf);
}
__device__ __forceinline__ v16bf fragB(const bf16_t* base, int klo) {
  return make_frag(base + klo / 2, base + klo / 2 + 16);
}
__device__ __forceinline__ v8f wmma_bf16(v16bf a, v16bf b, v8f c) {
  return __builtin_amdgcn_wmma_f32_16x16x32_bf16(false, a, false, b, (short)0, c,
                                                 false, false);
}

// ---------- fp32 -> bf16 conversion (vectorized) ----------
__global__ void cvt_f32_bf16(const float* __restrict__ src,
                             bf16_t* __restrict__ dst, int n4) {
  int i = blockIdx.x * blockDim.x + threadIdx.x;
  if (i < n4) {
    float4 f = reinterpret_cast<const float4*>(src)[i];
    v4bf o;
    o[0] = (bf16_t)f.x; o[1] = (bf16_t)f.y;
    o[2] = (bf16_t)f.z; o[3] = (bf16_t)f.w;
    reinterpret_cast<v4bf*>(dst)[i] = o;
  }
}

// ---------- projection GEMM: C[M,N] = X[M,K] * W[N,K]^T ----------
// mode 0: out bf16 [b,h,l,dk]    (q, k)
// mode 1: out bf16 [b,h,dk,l]    (v transposed for P*V)
// mode 2: out fp32 [m, N]        (final output projection)
// K-loop unrolled x2 with structural ping/pong register double-buffering:
// no rotation copies; next fragments load while current 8 wmmas execute.
__global__ __launch_bounds__(256)
void proj_gemm(const bf16_t* __restrict__ X, const bf16_t* __restrict__ W,
               void* __restrict__ out, int mode) {
  const int lane   = threadIdx.x & 31;
  const int wave   = threadIdx.x >> 5;
  const int lane15 = lane & 15;
  const int hi16   = lane >> 4;
  const int khalf  = hi16 * 8;
  const int klo    = hi16 * 16;
  const int row_base = blockIdx.x * 128 + (wave >> 1) * 32;  // 4 wave-rows
  const int col_base = blockIdx.y * 128 + (wave & 1) * 64;   // 2 wave-cols

  v8f acc[2][4] = {};
  const bf16_t* a0p = X + (size_t)(row_base + lane15) * D_MODEL;
  const bf16_t* a1p = X + (size_t)(row_base + 16 + lane15) * D_MODEL;
  const bf16_t* w0p = W + (size_t)(col_base + lane15) * D_MODEL;

  // ping buffers: kk = 0
  v16bf pa0 = fragA(a0p, khalf);
  v16bf pa1 = fragA(a1p, khalf);
  v16bf pbw[4];
#pragma unroll
  for (int j = 0; j < 4; ++j) pbw[j] = fragB(w0p + (size_t)j * 16 * D_MODEL, klo);

  for (int kk = 0; kk < D_MODEL; kk += 64) {
    // ---- load pong (kk+32) while computing with ping (kk) ----
    v16bf qa0 = fragA(a0p + kk + 32, khalf);
    v16bf qa1 = fragA(a1p + kk + 32, khalf);
    v16bf qbw[4];
#pragma unroll
    for (int j = 0; j < 4; ++j)
      qbw[j] = fragB(w0p + (size_t)j * 16 * D_MODEL + kk + 32, klo);
#pragma unroll
    for (int j = 0; j < 4; ++j) {
      acc[0][j] = wmma_bf16(pa0, pbw[j], acc[0][j]);
      acc[1][j] = wmma_bf16(pa1, pbw[j], acc[1][j]);
    }
    // ---- load ping (kk+64) while computing with pong (kk+32) ----
    // (final iteration over-reads 64B into the adjacent ws buffer; unused)
    pa0 = fragA(a0p + kk + 64, khalf);
    pa1 = fragA(a1p + kk + 64, khalf);
#pragma unroll
    for (int j = 0; j < 4; ++j)
      pbw[j] = fragB(w0p + (size_t)j * 16 * D_MODEL + kk + 64, klo);
#pragma unroll
    for (int j = 0; j < 4; ++j) {
      acc[0][j] = wmma_bf16(qa0, qbw[j], acc[0][j]);
      acc[1][j] = wmma_bf16(qa1, qbw[j], acc[1][j]);
    }
  }

#pragma unroll
  for (int i = 0; i < 2; ++i) {
#pragma unroll
    for (int j = 0; j < 4; ++j) {
#pragma unroll
      for (int r = 0; r < 8; ++r) {
        int row = row_base + i * 16 + r + khalf;  // khalf == 8*(lane>=16)
        int col = col_base + j * 16 + lane15;
        float v = acc[i][j][r];
        if (mode == 2) {
          reinterpret_cast<float*>(out)[(size_t)row * D_MODEL + col] = v;
        } else {
          int bb = row >> 11, ll = row & (SEQ - 1);
          int hh = col >> 6,  dd = col & (DKHEAD - 1);
          bf16_t bv = (bf16_t)v;
          if (mode == 0)
            reinterpret_cast<bf16_t*>(out)
              [(((size_t)(bb * NHEADS + hh) * SEQ) + ll) * DKHEAD + dd] = bv;
          else
            reinterpret_cast<bf16_t*>(out)
              [(((size_t)(bb * NHEADS + hh) * DKHEAD) + dd) * SEQ + ll] = bv;
        }
      }
    }
  }
}

// ---------- flash attention: one wave = 16 query rows, 64-key blocks ----------
__global__ __launch_bounds__(256)
void attn_fa(const bf16_t* __restrict__ q, const bf16_t* __restrict__ k,
             const bf16_t* __restrict__ vT, bf16_t* __restrict__ outc) {
  __shared__ bf16_t Pbuf[8][16 * 64];  // per-wave P staging, 16KB total
  const int lane   = threadIdx.x & 31;
  const int wave   = threadIdx.x >> 5;
  const int lane15 = lane & 15;
  const int hi16   = lane >> 4;
  const int khalf  = hi16 * 8;
  const int klo    = hi16 * 16;
  const int h = blockIdx.y;
  const int b = blockIdx.z;
  const int qbase = blockIdx.x * 128 + wave * 16;

  const bf16_t* qh = q  + (size_t)(b * NHEADS + h) * SEQ * DKHEAD;
  const bf16_t* kh = k  + (size_t)(b * NHEADS + h) * SEQ * DKHEAD;
  const bf16_t* vh = vT + (size_t)(b * NHEADS + h) * DKHEAD * SEQ;

  const bf16_t* qrow = qh + (size_t)(qbase + lane15) * DKHEAD;
  v16bf aq0 = fragA(qrow, khalf);        // dk 0..31
  v16bf aq1 = fragA(qrow + 32, khalf);   // dk 32..63

  v8f acc[4] = {};
  float mi[8], li[8];
#pragma unroll
  for (int r = 0; r < 8; ++r) { mi[r] = -1e30f; li[r] = 0.f; }

  bf16_t* Pw = &Pbuf[wave][0];

  for (int j = 0; j < SEQ; j += 64) {
    // ---- preload all K-block B fragments (one big load clause) ----
    v16bf bk[4][2];
#pragma unroll
    for (int t = 0; t < 4; ++t) {
      const bf16_t* kb = kh + (size_t)(j + t * 16 + lane15) * DKHEAD;
      bk[t][0] = fragB(kb, klo);
      bk[t][1] = fragB(kb + 32, klo);
    }
    // ---- scores S = (q . k^T) / sqrt(dk), 16 x 64 ----
    v8f s[4];
#pragma unroll
    for (int t = 0; t < 4; ++t) {
      v8f z = {};
      z = wmma_bf16(aq0, bk[t][0], z);
      z = wmma_bf16(aq1, bk[t][1], z);
      s[t] = z;
    }
    // ---- early-issue V-block loads; latency hidden under softmax ----
    v16bf bv[4][2];
#pragma unroll
    for (int nt = 0; nt < 4; ++nt) {
      const bf16_t* vb = vh + (size_t)(nt * 16 + lane15) * SEQ + j;
      bv[nt][0] = fragB(vb, klo);
      bv[nt][1] = fragB(vb + 32, klo);
    }

#pragma unroll
    for (int t = 0; t < 4; ++t)
#pragma unroll
      for (int r = 0; r < 8; ++r) s[t][r] *= 0.125f;  // 1/sqrt(64)

    // ---- online softmax (row stats reduced across 16-lane halves) ----
#pragma unroll
    for (int r = 0; r < 8; ++r) {
      float mx = fmaxf(fmaxf(s[0][r], s[1][r]), fmaxf(s[2][r], s[3][r]));
#pragma unroll
      for (int off = 1; off < 16; off <<= 1)
        mx = fmaxf(mx, __shfl_xor(mx, off, 32));
      float mnew  = fmaxf(mi[r], mx);
      float alpha = __expf(mi[r] - mnew);
      float rs = 0.f;
#pragma unroll
      for (int t = 0; t < 4; ++t) {
        float p = __expf(s[t][r] - mnew);
        s[t][r] = p;
        rs += p;
      }
#pragma unroll
      for (int off = 1; off < 16; off <<= 1)
        rs += __shfl_xor(rs, off, 32);
      li[r] = li[r] * alpha + rs;
      mi[r] = mnew;
#pragma unroll
      for (int nt = 0; nt < 4; ++nt) acc[nt][r] *= alpha;
    }

    // ---- stage P (bf16, row-major 16x64) in LDS, reload as A operand ----
#pragma unroll
    for (int t = 0; t < 4; ++t)
#pragma unroll
      for (int r = 0; r < 8; ++r)
        Pw[(r + khalf) * 64 + t * 16 + lane15] = (bf16_t)s[t][r];
    asm volatile("s_wait_dscnt 0x0" ::: "memory");
    v16bf ap0 = fragA(Pw + lane15 * 64, khalf);       // keys j..j+31
    v16bf ap1 = fragA(Pw + lane15 * 64 + 32, khalf);  // keys j+32..j+63

    // ---- O += P * V  (V fragments already resident) ----
#pragma unroll
    for (int nt = 0; nt < 4; ++nt) {
      acc[nt] = wmma_bf16(ap0, bv[nt][0], acc[nt]);
      acc[nt] = wmma_bf16(ap1, bv[nt][1], acc[nt]);
    }
  }

  // ---- normalize and emit to concat [b, l, h*64+d] (bf16) ----
#pragma unroll
  for (int r = 0; r < 8; ++r) {
    float inv = 1.f / li[r];
    int row = qbase + r + khalf;
#pragma unroll
    for (int nt = 0; nt < 4; ++nt) {
      float v = acc[nt][r] * inv;
      outc[(size_t)(b * SEQ + row) * D_MODEL + h * DKHEAD + nt * 16 + lane15] =
          (bf16_t)v;
    }
  }
}

extern "C" void kernel_launch(void* const* d_in, const int* in_sizes, int n_in,
                              void* d_out, int out_size, void* d_ws, size_t ws_size,
                              hipStream_t stream) {
  (void)in_sizes; (void)n_in; (void)out_size; (void)ws_size;
  const float* Q  = (const float*)d_in[0];
  const float* K  = (const float*)d_in[1];
  const float* V  = (const float*)d_in[2];
  const float* WQ = (const float*)d_in[3];
  const float* WK = (const float*)d_in[4];
  const float* WV = (const float*)d_in[5];
  const float* WO = (const float*)d_in[6];

  const size_t NX = (size_t)MROWS * D_MODEL;     // 8M elems
  const size_t NW = (size_t)D_MODEL * D_MODEL;   // 1M elems
  bf16_t* Xq = (bf16_t*)d_ws;
  bf16_t* Xk = Xq + NX;
  bf16_t* Xv = Xk + NX;
  bf16_t* wq = Xv + NX;
  bf16_t* wk = wq + NW;
  bf16_t* wv = wk + NW;
  bf16_t* wo = wv + NW;
  bf16_t* qh = wo + NW;
  bf16_t* kh = qh + NX;
  bf16_t* vT = kh + NX;
  bf16_t* concat = Xq;  // Xq is dead after the q projection

  const int nx4 = (int)(NX / 4), nw4 = (int)(NW / 4);
  cvt_f32_bf16<<<(nx4 + 255) / 256, 256, 0, stream>>>(Q,  Xq, nx4);
  cvt_f32_bf16<<<(nx4 + 255) / 256, 256, 0, stream>>>(K,  Xk, nx4);
  cvt_f32_bf16<<<(nx4 + 255) / 256, 256, 0, stream>>>(V,  Xv, nx4);
  cvt_f32_bf16<<<(nw4 + 255) / 256, 256, 0, stream>>>(WQ, wq, nw4);
  cvt_f32_bf16<<<(nw4 + 255) / 256, 256, 0, stream>>>(WK, wk, nw4);
  cvt_f32_bf16<<<(nw4 + 255) / 256, 256, 0, stream>>>(WV, wv, nw4);
  cvt_f32_bf16<<<(nw4 + 255) / 256, 256, 0, stream>>>(WO, wo, nw4);

  dim3 pgrid(MROWS / 128, D_MODEL / 128);
  proj_gemm<<<pgrid, 256, 0, stream>>>(Xq, wq, qh, 0);
  proj_gemm<<<pgrid, 256, 0, stream>>>(Xk, wk, kh, 0);
  proj_gemm<<<pgrid, 256, 0, stream>>>(Xv, wv, vT, 1);

  attn_fa<<<dim3(SEQ / 128, NHEADS, BATCH), 256, 0, stream>>>(qh, kh, vT, concat);

  proj_gemm<<<pgrid, 256, 0, stream>>>(concat, wo, d_out, 2);
}
